// PointNet2MSG_16054587752589
// MI455X (gfx1250) — matrix operand order repaired and checked by reading, not compiled, not run on hardware
//
#include <hip/hip_runtime.h>
#include <hip/hip_bf16.h>

typedef __attribute__((ext_vector_type(16))) _Float16 v16h;
typedef __attribute__((ext_vector_type(8)))  _Float16 v8h;
typedef __attribute__((ext_vector_type(8)))  float    v8f;
typedef __attribute__((ext_vector_type(4)))  int      v4i_;

static inline int pad16(int x){ return (x+15)&~15; }
static inline int pad32(int x){ return (x+31)&~31; }
static inline int pad64(int x){ return (x+63)&~63; }
static inline long cdivl(long a, long b){ return (a+b-1)/b; }

static __device__ __forceinline__ int clampi(int v, int lo, int hi){
  return v < lo ? lo : (v > hi ? hi : v);
}

// ---------------------------------------------------------------------------
// CDNA5 async global->LDS staging (ASYNCcnt path), with sync fallback.
// Builtin signature (from compiler diagnostic): takes global `int4*` src.
// ---------------------------------------------------------------------------
#if defined(__has_builtin)
#  if __has_builtin(__builtin_amdgcn_global_load_async_to_lds_b128)
#    define USE_ASYNC_LDS 1
#  endif
#endif
#ifndef USE_ASYNC_LDS
#  define USE_ASYNC_LDS 0
#endif

typedef __attribute__((address_space(1))) v4i_ gv4i;
typedef __attribute__((address_space(3))) v4i_ lv4i;

static __device__ __forceinline__ void g2l_b128(const _Float16* g, _Float16* l) {
#if USE_ASYNC_LDS
  __builtin_amdgcn_global_load_async_to_lds_b128((gv4i*)g, (lv4i*)l, 0, 0);
#else
  *(v8h*)l = *(const v8h*)g;
#endif
}

static __device__ __forceinline__ void g2l_wait() {
#if USE_ASYNC_LDS
  asm volatile("s_wait_asynccnt 0x0" ::: "memory");
#endif
}

// ---------------------------------------------------------------------------
// Core WMMA GEMM:  Om[n, o] = sum_c Wm[o, c] * Xm[n, c]
// Wm: f16 [Op, Cp] row-major (padded), Xm: f16 [Ntot, Cp] row-major,
// Om: f32 [Ntot, Op]. Cp % 32 == 0, Op % 16 == 0, Ntot % 64 == 0.
// Block = 128 threads = 4 waves computing a 64(M) x 64(N) tile.
// The 64x32 B chunk is staged to LDS (double buffered, async on CDNA5) and
// shared by all 4 waves; each wave holds one A tile and 4 f32 accumulators.
// ---------------------------------------------------------------------------
#define SB_STRIDE 40   // halves per LDS row: 80B, 16B aligned, conflict-free

__global__ __launch_bounds__(128) void k_gemm(const _Float16* __restrict__ Wm,
                                              const _Float16* __restrict__ Xm,
                                              float* __restrict__ Om,
                                              int Cp, int Op, int Ntot) {
  __shared__ _Float16 sb[2][64 * SB_STRIDE];
  const int tid  = threadIdx.x;
  const int lane = tid & 31;
  const int wave = tid >> 5;            // 0..3
  const int hi   = lane >> 4;           // half-wave select
  const int l16  = lane & 15;
  const int nb   = blockIdx.y * 64;
  const int mo   = blockIdx.x * 64 + wave * 16;
  const bool active = (mo < Op);        // wave-uniform

  // staging map: 2 x b128 per thread covers 64 rows x 32 halves
  const int r0 = tid >> 2;              // 0..31
  const int r1 = r0 + 32;               // 32..63
  const int c0 = (tid & 3) * 8;         // 0,8,16,24 halves
  const _Float16* xr0 = Xm + (size_t)(nb + r0) * Cp + c0;
  const _Float16* xr1 = Xm + (size_t)(nb + r1) * Cp + c0;
  const int l0 = r0 * SB_STRIDE + c0;
  const int l1 = r1 * SB_STRIDE + c0;

  // prologue: stage K-chunk 0 into buffer 0
  g2l_b128(xr0, &sb[0][l0]);
  g2l_b128(xr1, &sb[0][l1]);
  g2l_wait();
  __syncthreads();

  const _Float16* wrow = Wm + (size_t)(mo + l16) * Cp;
  const int ab = hi * 8;   // A: lanes 0-15 K 0..7 & 16..23, lanes 16-31 K 8..15 & 24..31
  const int bb = hi * 8;   // B: lanes 0-15 K 0..7 & 16..23, lanes 16-31 K 8..15 & 24..31
  v8f acc[4] = {};
  int cur = 0;
  for (int kc = 0; kc < Cp; kc += 32) {
    if (kc + 32 < Cp) {  // stage next chunk while computing (async DMA in flight)
      g2l_b128(xr0 + kc + 32, &sb[cur ^ 1][l0]);
      g2l_b128(xr1 + kc + 32, &sb[cur ^ 1][l1]);
    }
    if (active) {
      v8h a0 = *(const v8h*)(wrow + kc + ab);
      v8h a1 = *(const v8h*)(wrow + kc + 16 + ab);
      __builtin_prefetch(wrow + kc + 64, 0, 0);
      v16h av;
#pragma unroll
      for (int q = 0; q < 8; ++q) { av[q] = a0[q]; av[q + 8] = a1[q]; }
#pragma unroll
      for (int t = 0; t < 4; ++t) {
        const _Float16* lr = &sb[cur][(t * 16 + l16) * SB_STRIDE + bb];
        v8h b0 = *(const v8h*)lr;
        v8h b1 = *(const v8h*)(lr + 16);
        v16h bv;
#pragma unroll
        for (int q = 0; q < 8; ++q) { bv[q] = b0[q]; bv[q + 8] = b1[q]; }
        acc[t] = __builtin_amdgcn_wmma_f32_16x16x32_f16(false, av, false, bv,
                                                        (short)0, acc[t], false, false);
      }
    }
    g2l_wait();
    __syncthreads();
    cur ^= 1;
  }
  if (active) {
#pragma unroll
    for (int t = 0; t < 4; ++t) {
      // C layout: VGPR r -> (M = r + hi*8, N = l16)
      float* orow = Om + (size_t)(nb + t * 16 + l16) * Op + mo + hi * 8;
      *(v8f*)orow = acc[t];
    }
  }
}

// Pad/convert f32 weights [O, C] -> f16 [Op, Cp]
__global__ void k_padw(const float* __restrict__ W, _Float16* __restrict__ Wp,
                       int O, int C, int Op, int Cp) {
  long t = (long)blockIdx.x * blockDim.x + threadIdx.x;
  long tot = (long)Op * Cp; if (t >= tot) return;
  int c = t % Cp; int o = t / Cp;
  float v = (o < O && c < C) ? W[(size_t)o * C + c] : 0.f;
  Wp[t] = (_Float16)v;
}

// Per-channel batch-norm statistics over valid rows of both batches.
__global__ void k_bnstats(const float* __restrict__ X, int Op, int rowsPad, int rowsValid,
                          int B, float* __restrict__ stats) {
  int o = blockIdx.x; int tid = threadIdx.x;
  __shared__ float ss[256]; __shared__ float sq[256];
  float s = 0.f, q = 0.f;
  for (int b = 0; b < B; ++b) {
    const float* base = X + (size_t)b * rowsPad * Op + o;
    for (int r = tid; r < rowsValid; r += 256) {
      float v = base[(size_t)r * Op]; s += v; q += v * v;
    }
  }
  ss[tid] = s; sq[tid] = q; __syncthreads();
  for (int k = 128; k > 0; k >>= 1) {
    if (tid < k) { ss[tid] += ss[tid + k]; sq[tid] += sq[tid + k]; }
    __syncthreads();
  }
  if (tid == 0) {
    float cnt = (float)B * rowsValid;
    float m = ss[0] / cnt;
    float v = sq[0] / cnt - m * m;
    stats[o] = m; stats[Op + o] = rsqrtf(v + 1e-5f);
  }
}

// y = relu(g*(x-mean)*rstd + beta), f32 [rows,Op] -> f16 [rows,Cpo], pad cols zeroed
__global__ void k_bnrelu(const float* __restrict__ X, const float* __restrict__ stats,
                         const float* __restrict__ g, const float* __restrict__ bt,
                         _Float16* __restrict__ Y, int Op, int O, int Cpo, long rows) {
  long t = (long)blockIdx.x * blockDim.x + threadIdx.x;
  long tot = rows * Cpo; if (t >= tot) return;
  int c = t % Cpo; long r = t / Cpo;
  float y = 0.f;
  if (c < O) {
    float x = X[r * Op + c];
    y = g[c] * (x - stats[c]) * stats[Op + c] + bt[c];
    y = fmaxf(y, 0.f);
  }
  Y[t] = (_Float16)y;
}

// plain f32 [rows,Op] -> f16 [rows,Cpo]
__global__ void k_cvt16(const float* __restrict__ X, _Float16* __restrict__ Y,
                        int Op, int O, int Cpo, long rows) {
  long t = (long)blockIdx.x * blockDim.x + threadIdx.x;
  long tot = rows * Cpo; if (t >= tot) return;
  int c = t % Cpo; long r = t / Cpo;
  Y[t] = (_Float16)(c < O ? X[r * Op + c] : 0.f);
}

// ---------------------------------------------------------------------------
// Farthest point sampling (one block per batch)
// ---------------------------------------------------------------------------
__global__ void k_fps(const float* __restrict__ xyz, int N, int S,
                      int* __restrict__ idx, float* __restrict__ dmin) {
  int b = blockIdx.x; int tid = threadIdx.x;
  __shared__ float sv[256]; __shared__ int si[256]; __shared__ int lastI;
  const float* p = xyz + (size_t)b * N * 3;
  for (int j = tid; j < N; j += 256) dmin[(size_t)b * N + j] = 1e10f;
  if (tid == 0) { idx[(size_t)b * S] = 0; lastI = 0; }
  __syncthreads();
  for (int it = 1; it < S; ++it) {
    int last = lastI;
    float lx = p[last * 3 + 0], ly = p[last * 3 + 1], lz = p[last * 3 + 2];
    float best = -1.f; int bi = 0;
    for (int j = tid; j < N; j += 256) {
      float dx = p[j * 3 + 0] - lx, dy = p[j * 3 + 1] - ly, dz = p[j * 3 + 2] - lz;
      float d = dx * dx + dy * dy + dz * dz;
      float m = dmin[(size_t)b * N + j];
      m = fminf(m, d); dmin[(size_t)b * N + j] = m;
      if (m > best) { best = m; bi = j; }
    }
    sv[tid] = best; si[tid] = bi; __syncthreads();
    for (int k = 128; k > 0; k >>= 1) {
      if (tid < k && sv[tid + k] > sv[tid]) { sv[tid] = sv[tid + k]; si[tid] = si[tid + k]; }
      __syncthreads();
    }
    if (tid == 0) { idx[(size_t)b * S + it] = si[0]; lastI = si[0]; }
    __syncthreads();
  }
}

__global__ void k_gather_rows(const float* __restrict__ src, const int* __restrict__ idx,
                              float* __restrict__ dst, int cols, int S, int Nsrc, int B) {
  long t = (long)blockIdx.x * blockDim.x + threadIdx.x;
  long tot = (long)B * S * cols; if (t >= tot) return;
  int c = t % cols; long r = t / cols; int s = r % S; int b = r / S;
  int j = idx[(size_t)b * S + s];
  dst[t] = src[((size_t)b * Nsrc + j) * cols + c];
}

__global__ void k_ballq(const float* __restrict__ xyz, const float* __restrict__ nxyz,
                        int* __restrict__ out, int N, int S, int K, float r2, int B) {
  long t = (long)blockIdx.x * blockDim.x + threadIdx.x;
  if (t >= (long)B * S) return;
  int b = t / S, s = t % S;
  const float* q = nxyz + ((size_t)b * S + s) * 3;
  const float* p = xyz + (size_t)b * N * 3;
  int* o = out + (size_t)t * K;
  int cnt = 0; int first = N - 1;
  for (int j = 0; j < N && cnt < K; ++j) {
    float dx = p[j * 3 + 0] - q[0], dy = p[j * 3 + 1] - q[1], dz = p[j * 3 + 2] - q[2];
    float d = dx * dx + dy * dy + dz * dz;
    if (d < r2) { if (cnt == 0) first = j; o[cnt++] = j; }
  }
  for (int k = cnt; k < K; ++k) o[k] = first;
}

// grouped row (b, s*K+k): [dx,dy,dz, feat(src)...] into f16 [B*S*K, Cp]
__global__ void k_group(const float* __restrict__ xyz, const float* __restrict__ nxyz,
                        const int* __restrict__ gidx, const _Float16* __restrict__ feat,
                        int N, int S, int K, int Cf, int CfP, int featRows,
                        int Cp, _Float16* __restrict__ out, int B) {
  long t = (long)blockIdx.x * blockDim.x + threadIdx.x;
  long tot = (long)B * S * K; if (t >= tot) return;
  int b = t / (S * K); int sk = t % (S * K); int s = sk / K;
  int src = gidx[t];
  const float* q = nxyz + ((size_t)b * S + s) * 3;
  const float* p = xyz + ((size_t)b * N + src) * 3;
  _Float16* row = out + (size_t)t * Cp;
  row[0] = (_Float16)(p[0] - q[0]);
  row[1] = (_Float16)(p[1] - q[1]);
  row[2] = (_Float16)(p[2] - q[2]);
  if (Cf > 0) {
    const _Float16* f = feat + ((size_t)b * featRows + src) * CfP;
    for (int c = 0; c < Cf; ++c) row[3 + c] = f[c];
  }
}

__global__ void k_maxpool(const _Float16* __restrict__ X, int S, int K, int O, int Cp,
                          _Float16* __restrict__ out, int Spad, int Cpo, int off, int B) {
  long t = (long)blockIdx.x * blockDim.x + threadIdx.x;
  long tot = (long)B * S * O; if (t >= tot) return;
  int o = t % O; long r = t / O; int s = r % S; int b = r / S;
  const _Float16* x = X + ((size_t)(b * S + s)) * K * Cp + o;
  float m = -1e30f;
  for (int k = 0; k < K; ++k) m = fmaxf(m, (float)x[(size_t)k * Cp]);
  out[((size_t)b * Spad + s) * Cpo + off + o] = (_Float16)m;
}

// bilinear sample of XT-layout image features
__global__ void k_bilinear(const _Float16* __restrict__ img, int H, int W, int RpadIn,
                           const float* __restrict__ xy, int S, int Spad, int C, int Cp,
                           _Float16* __restrict__ out, int B) {
  long t = (long)blockIdx.x * blockDim.x + threadIdx.x;
  if (t >= (long)B * S) return;
  int b = t / S, s = t % S;
  float xv = xy[((size_t)b * S + s) * 2 + 0];
  float yv = xy[((size_t)b * S + s) * 2 + 1];
  float x = (xv + 1.f) * 0.5f * (W - 1);
  float y = (yv + 1.f) * 0.5f * (H - 1);
  float x0 = floorf(x), y0 = floorf(y);
  float wx = x - x0, wy = y - y0;
  int xi0 = clampi((int)x0, 0, W - 1), xi1 = clampi((int)x0 + 1, 0, W - 1);
  int yi0 = clampi((int)y0, 0, H - 1), yi1 = clampi((int)y0 + 1, 0, H - 1);
  const _Float16* r00 = img + ((size_t)b * RpadIn + (size_t)yi0 * W + xi0) * Cp;
  const _Float16* r01 = img + ((size_t)b * RpadIn + (size_t)yi0 * W + xi1) * Cp;
  const _Float16* r10 = img + ((size_t)b * RpadIn + (size_t)yi1 * W + xi0) * Cp;
  const _Float16* r11 = img + ((size_t)b * RpadIn + (size_t)yi1 * W + xi1) * Cp;
  float w00 = (1.f - wy) * (1.f - wx), w01 = (1.f - wy) * wx;
  float w10 = wy * (1.f - wx), w11 = wy * wx;
  _Float16* o = out + ((size_t)b * Spad + s) * Cp;
  for (int c = 0; c < C; ++c)
    o[c] = (_Float16)(w00 * (float)r00[c] + w01 * (float)r01[c] +
                      w10 * (float)r10[c] + w11 * (float)r11[c]);
}

// att = sigmoid( fc3 . tanh(ri + b1 + rp + b2) + b3 )
__global__ void k_att(const float* __restrict__ ri, const float* __restrict__ rp,
                      const float* __restrict__ b1, const float* __restrict__ b2,
                      const float* __restrict__ w3, const float* __restrict__ b3,
                      int rc, int Op, float* __restrict__ att, long rows) {
  long t = (long)blockIdx.x * blockDim.x + threadIdx.x;
  if (t >= rows) return;
  const float* a = ri + (size_t)t * Op;
  const float* b = rp + (size_t)t * Op;
  float acc = 0.f;
  for (int r = 0; r < rc; ++r) acc += w3[r] * tanhf(a[r] + b1[r] + b[r] + b2[r]);
  att[t] = 1.f / (1.f + expf(-(acc + b3[0])));
}

__global__ void k_scale_rows(_Float16* __restrict__ Y, const float* __restrict__ att,
                             int Cp, long rows) {
  long t = (long)blockIdx.x * blockDim.x + threadIdx.x;
  if (t >= rows) return;
  float a = att[t];
  _Float16* y = Y + (size_t)t * Cp;
  for (int c = 0; c < Cp; ++c) y[c] = (_Float16)((float)y[c] * a);
}

__global__ void k_copycols16(const _Float16* __restrict__ X, int Cpi,
                             _Float16* __restrict__ Y, int Cpo, int off, int Ccopy, long rows) {
  long t = (long)blockIdx.x * blockDim.x + threadIdx.x;
  long tot = rows * Ccopy; if (t >= tot) return;
  int c = t % Ccopy; long r = t / Ccopy;
  Y[r * Cpo + off + c] = X[r * Cpi + c];
}

// 3-NN inverse-distance interpolation of f16 features
__global__ void k_interp3(const float* __restrict__ ux, const float* __restrict__ kx,
                          int Nu, int Sk, const _Float16* __restrict__ f2, int f2rows,
                          int C2, int C2p, _Float16* __restrict__ Y, int Cpo, int off, int B) {
  long t = (long)blockIdx.x * blockDim.x + threadIdx.x;
  if (t >= (long)B * Nu) return;
  int b = t / Nu, n = t % Nu;
  const float* u = ux + ((size_t)b * Nu + n) * 3;
  const float* kp = kx + (size_t)b * Sk * 3;
  float d0 = 1e30f, d1 = 1e30f, d2 = 1e30f; int i0 = 0, i1 = 0, i2 = 0;
  for (int j = 0; j < Sk; ++j) {
    float dx = u[0] - kp[j * 3 + 0], dy = u[1] - kp[j * 3 + 1], dz = u[2] - kp[j * 3 + 2];
    float d = dx * dx + dy * dy + dz * dz;
    if (d < d0) { d2 = d1; i2 = i1; d1 = d0; i1 = i0; d0 = d; i0 = j; }
    else if (d < d1) { d2 = d1; i2 = i1; d1 = d; i1 = j; }
    else if (d < d2) { d2 = d; i2 = j; }
  }
  float w0 = 1.f / (d0 + 1e-8f), w1 = 1.f / (d1 + 1e-8f), w2 = 1.f / (d2 + 1e-8f);
  float ws = w0 + w1 + w2; w0 /= ws; w1 /= ws; w2 /= ws;
  const _Float16* r0 = f2 + ((size_t)b * f2rows + i0) * C2p;
  const _Float16* r1 = f2 + ((size_t)b * f2rows + i1) * C2p;
  const _Float16* r2 = f2 + ((size_t)b * f2rows + i2) * C2p;
  _Float16* y = Y + ((size_t)b * Nu + n) * Cpo + off;
  for (int c = 0; c < C2; ++c)
    y[c] = (_Float16)(w0 * (float)r0[c] + w1 * (float)r1[c] + w2 * (float)r2[c]);
}

// im2col (3x3, pad 1) from XT f16 image to [rows, C*9 padded]
__global__ void k_im2col(const _Float16* __restrict__ X, int Hi, int Wi, int RpadIn,
                         int Ci, int CpIn, _Float16* __restrict__ Y,
                         int Ho, int Wo, int RpadOut, int C9p, int stride, int B) {
  long t = (long)blockIdx.x * blockDim.x + threadIdx.x;
  long tot = (long)B * Ho * Wo * Ci; if (t >= tot) return;
  int cch = t % Ci; long r2 = t / Ci; int ro = r2 % (Ho * Wo); int b = r2 / (Ho * Wo);
  int ho = ro / Wo, wo = ro % Wo;
  _Float16* y = Y + ((size_t)b * RpadOut + ro) * C9p + cch * 9;
  for (int ky = 0; ky < 3; ++ky)
    for (int kx = 0; kx < 3; ++kx) {
      int yi = ho * stride + ky - 1, xi = wo * stride + kx - 1;
      _Float16 v = (_Float16)0.0f;
      if (yi >= 0 && yi < Hi && xi >= 0 && xi < Wi)
        v = X[((size_t)b * RpadIn + (size_t)yi * Wi + xi) * CpIn + cch];
      y[ky * 3 + kx] = v;
    }
}

__global__ void k_img_to_xt(const float* __restrict__ img, _Float16* __restrict__ Y,
                            int HW, int Rpad, int C, int Cp, int B) {
  long t = (long)blockIdx.x * blockDim.x + threadIdx.x;
  long tot = (long)B * HW * C; if (t >= tot) return;
  int c = t % C; long r = t / C; int p = r % HW; int b = r / HW;
  Y[((size_t)b * Rpad + p) * Cp + c] = (_Float16)img[((size_t)b * C + c) * HW + p];
}

// deconv upsample: 16 output channels per source block, written into 64-ch concat buffer
__global__ void k_deconv(const _Float16* __restrict__ X, int Hi, int Wi, int RpadIn,
                         int Ci, int CpIn, const float* __restrict__ Wd,
                         const float* __restrict__ bd, int K,
                         _Float16* __restrict__ Y, int RpadOut, int ocOff, int B) {
  long t = (long)blockIdx.x * blockDim.x + threadIdx.x;
  int Wf = Wi * K;
  long HWf = (long)Hi * K * Wf;
  long tot = (long)B * HWf * 16; if (t >= tot) return;
  int o = t % 16; long r2 = t / 16; long ro = r2 % HWf; int b = r2 / HWf;
  int hp = ro / Wf, wq = ro % Wf;
  int h = hp / K, p = hp % K, w = wq / K, q = wq % K;
  const _Float16* xr = X + ((size_t)b * RpadIn + (size_t)h * Wi + w) * CpIn;
  float acc = bd[o];
  for (int c = 0; c < Ci; ++c)
    acc += (float)xr[c] * Wd[(((size_t)c * 16 + o) * K + p) * K + q];
  Y[((size_t)b * RpadOut + ro) * 64 + ocOff + o] = (_Float16)acc;
}

__global__ void k_copyf(const float* __restrict__ src, float* __restrict__ dst, long n) {
  long t = (long)blockIdx.x * blockDim.x + threadIdx.x;
  if (t < n) dst[t] = src[t];
}

// XT f16 [B*Npts, Cp] -> f32 [B, C, Npts]
__global__ void k_out_bcn(const _Float16* __restrict__ X, float* __restrict__ dst,
                          int Npts, int C, int Cp, int B) {
  long t = (long)blockIdx.x * blockDim.x + threadIdx.x;
  long tot = (long)B * C * Npts; if (t >= tot) return;
  int n = t % Npts; long r = t / Npts; int o = r % C; int b = r / C;
  dst[t] = (float)X[((size_t)b * Npts + n) * Cp + o];
}

// ---------------------------------------------------------------------------
// Host-side orchestration helpers
// ---------------------------------------------------------------------------
struct Ctx {
  hipStream_t st;
  float* fbuf; float* fbuf2; float* stats;
  _Float16* wbuf;
};

struct AttnP {
  const float *bn_b, *bn_g, *conv_b, *conv_w;
  const float *ia_bn_b, *ia_bn_g, *ia_conv_b, *ia_conv_w;
  const float *fc1_b, *fc1_w, *fc2_b, *fc2_w, *fc3_b, *fc3_w;
};

static int h_gemm(const Ctx& c, const float* W, int O, int Cin,
                  const _Float16* X, int Cp, int Ntot, float* out) {
  int Op = pad16(O);
  long tw = (long)Op * Cp;
  k_padw<<<(int)cdivl(tw, 256), 256, 0, c.st>>>(W, c.wbuf, O, Cin, Op, Cp);
  dim3 g((unsigned)cdivl(Op, 64), (unsigned)(Ntot / 64));
  k_gemm<<<g, 128, 0, c.st>>>(c.wbuf, X, out, Cp, Op, Ntot);
  return Op;
}

static void h_bnrelu(const Ctx& c, const float* Xf, int Op, int O, const float* g,
                     const float* bt, _Float16* Y, int rowsPad, int rowsValid, int B) {
  k_bnstats<<<Op, 256, 0, c.st>>>(Xf, Op, rowsPad, rowsValid, B, c.stats);
  int Cpo = pad32(O);
  long tot = (long)B * rowsPad * Cpo;
  k_bnrelu<<<(int)cdivl(tot, 256), 256, 0, c.st>>>(Xf, c.stats, g, bt, Y, Op, O, Cpo,
                                                   (long)B * rowsPad);
}

static void h_layer(const Ctx& c, const float* W, const float* g, const float* bt,
                    int O, int Cin, const _Float16* X, int Cp,
                    _Float16* Y, int rowsPad, int rowsValid, int B) {
  int Op = h_gemm(c, W, O, Cin, X, Cp, B * rowsPad, c.fbuf);
  h_bnrelu(c, c.fbuf, Op, O, g, bt, Y, rowsPad, rowsValid, B);
}

static void h_convblock(const Ctx& c, const float* w1, const float* g1, const float* b1,
                        const float* w2, int Ci, int Co, int Hi, int Wi,
                        const _Float16* Xin, int CpIn,
                        _Float16* ping, _Float16* pong, _Float16* imgOut, int B) {
  int HWi = Hi * Wi, Ri = pad64(HWi);
  int C9 = Ci * 9, C9p = pad32(C9);
  (void)hipMemsetAsync(ping, 0, (size_t)B * Ri * C9p * sizeof(_Float16), c.st);
  long t1 = (long)B * HWi * Ci;
  k_im2col<<<(int)cdivl(t1, 256), 256, 0, c.st>>>(Xin, Hi, Wi, Ri, Ci, CpIn,
                                                  ping, Hi, Wi, Ri, C9p, 1, B);
  int Op = h_gemm(c, w1, Co, C9, ping, C9p, B * Ri, c.fbuf);
  h_bnrelu(c, c.fbuf, Op, Co, g1, b1, pong, Ri, HWi, B);
  int Ho = Hi / 2, Wo = Wi / 2, HWo = Ho * Wo, Ro = pad64(HWo);
  int CoP = pad32(Co);
  int C9b = Co * 9, C9bp = pad32(C9b);
  (void)hipMemsetAsync(ping, 0, (size_t)B * Ro * C9bp * sizeof(_Float16), c.st);
  long t2 = (long)B * HWo * Co;
  k_im2col<<<(int)cdivl(t2, 256), 256, 0, c.st>>>(pong, Hi, Wi, Ri, Co, CoP,
                                                  ping, Ho, Wo, Ro, C9bp, 2, B);
  Op = h_gemm(c, w2, Co, C9b, ping, C9bp, B * Ro, c.fbuf);
  long t3 = (long)B * Ro * CoP;
  k_cvt16<<<(int)cdivl(t3, 256), 256, 0, c.st>>>(c.fbuf, imgOut, Op, Co, CoP, (long)B * Ro);
}

static void h_attention(const Ctx& c, const AttnP& a, int pcc, int Cimg,
                        int Spad, int Sv, int B, const _Float16* pointF,
                        const _Float16* imgF, _Float16* imgf_t, _Float16* cat_t,
                        float* att, _Float16* outF) {
  int rc = pcc / 4, rcp = pad16(rc), CimgP = pad32(Cimg);
  int rows = B * Spad;
  h_gemm(c, a.fc1_w, rc, Cimg, imgF, CimgP, rows, c.fbuf);
  h_gemm(c, a.fc2_w, rc, pcc, pointF, pcc, rows, c.fbuf2);
  k_att<<<(int)cdivl(rows, 256), 256, 0, c.st>>>(c.fbuf, c.fbuf2, a.fc1_b, a.fc2_b,
                                                 a.fc3_w, a.fc3_b, rc, rcp, att, rows);
  int Op = h_gemm(c, a.ia_conv_w, pcc, Cimg, imgF, CimgP, rows, c.fbuf);
  h_bnrelu(c, c.fbuf, Op, pcc, a.ia_bn_g, a.ia_bn_b, imgf_t, Spad, Sv, B);
  k_scale_rows<<<(int)cdivl(rows, 256), 256, 0, c.st>>>(imgf_t, att, pcc, rows);
  long tc = (long)rows * pcc;
  k_copycols16<<<(int)cdivl(tc, 256), 256, 0, c.st>>>(pointF, pcc, cat_t, 2 * pcc, 0, pcc, rows);
  k_copycols16<<<(int)cdivl(tc, 256), 256, 0, c.st>>>(imgf_t, pcc, cat_t, 2 * pcc, pcc, pcc, rows);
  Op = h_gemm(c, a.conv_w, pcc, 2 * pcc, cat_t, 2 * pcc, rows, c.fbuf);
  h_bnrelu(c, c.fbuf, Op, pcc, a.bn_g, a.bn_b, outF, Spad, Sv, B);
}

// ---------------------------------------------------------------------------
extern "C" void kernel_launch(void* const* d_in, const int* in_sizes, int n_in,
                              void* d_out, int out_size, void* d_ws, size_t ws_size,
                              hipStream_t stream) {
  (void)in_sizes; (void)n_in; (void)out_size; (void)ws_size;
  const int B = 2, N0 = 8192, HH = 128, WW = 416;
  static const int NPTS[4] = {2048, 512, 128, 32};
  static const float RAD2[4][2] = {{0.01f, 0.25f}, {0.25f, 1.f}, {1.f, 4.f}, {4.f, 16.f}};
  static const int NS[2] = {16, 32};
  static const int MLP[4][2][3] = {
      {{16, 16, 32}, {32, 32, 64}},
      {{64, 64, 128}, {64, 96, 128}},
      {{128, 196, 256}, {128, 196, 256}},
      {{256, 256, 512}, {256, 384, 512}}};
  static const int PCC[4] = {96, 256, 512, 1024};
  static const int IMGC[5] = {3, 64, 128, 256, 512};
  static const int FPM[4][2] = {{128, 128}, {256, 256}, {512, 512}, {512, 512}};
  static const int DK[4] = {2, 4, 8, 16};
  static const int HS[4] = {128, 64, 32, 16};
  static const int WS_[4] = {416, 208, 104, 52};

  const float* pc    = (const float*)d_in[0];
  const float* image = (const float*)d_in[1];
  const float* xy    = (const float*)d_in[2];

  // -------- parameter unflattening (jax pytree order: sorted dict keys) ------
  int pi = 3;
  auto P = [&]() -> const float* { return (const float*)d_in[pi++]; };
  const float *dec_b[4], *dec_w[4];
  for (int j = 0; j < 4; ++j) { dec_b[j] = P(); dec_w[j] = P(); }
  auto readAttn = [&](AttnP& a) {
    a.bn_b = P(); a.bn_g = P(); a.conv_b = P(); a.conv_w = P();
    a.ia_bn_b = P(); a.ia_bn_g = P(); a.ia_conv_b = P(); a.ia_conv_w = P();
    a.fc1_b = P(); a.fc1_w = P(); a.fc2_b = P(); a.fc2_w = P();
    a.fc3_b = P(); a.fc3_w = P();
  };
  AttnP ffp; readAttn(ffp);
  const float *fpW[4][2], *fpG[4][2], *fpB[4][2];
  for (int k = 0; k < 4; ++k)
    for (int l = 0; l < 2; ++l) { fpW[k][l] = P(); fpG[k][l] = P(); fpB[k][l] = P(); }
  AttnP fus[4];
  for (int i = 0; i < 4; ++i) readAttn(fus[i]);
  const float *ib_b1[4], *ib_g1[4], *ib_w1[4], *ib_w2[4];
  for (int i = 0; i < 4; ++i) { ib_b1[i] = P(); ib_g1[i] = P(); ib_w1[i] = P(); ib_w2[i] = P(); }
  const float* ifc_b = P(); const float* ifc_beta = P();
  const float* ifc_g = P(); const float* ifc_w = P();
  const float *saW[4][2][3], *saG[4][2][3], *saB[4][2][3];
  for (int i = 0; i < 4; ++i)
    for (int j = 0; j < 2; ++j)
      for (int l = 0; l < 3; ++l) { saW[i][j][l] = P(); saG[i][j][l] = P(); saB[i][j][l] = P(); }

  // -------- workspace bump allocator ----------------------------------------
  size_t wsoff = 0;
  auto A = [&](size_t bytes) -> void* {
    void* p = (void*)((char*)d_ws + wsoff);
    wsoff = (wsoff + bytes + 255) & ~(size_t)255;
    return p;
  };
  float*    fbuf  = (float*)A((size_t)40 << 20);
  float*    fbuf2 = (float*)A((size_t)4 << 20);
  _Float16* ping  = (_Float16*)A((size_t)40 << 20);
  _Float16* pong  = (_Float16*)A((size_t)40 << 20);
  _Float16* wbuf  = (_Float16*)A((size_t)8 << 20);
  float*    stats = (float*)A(4096 * sizeof(float));
  int*      fidx  = (int*)A((size_t)B * 2048 * sizeof(int));
  int*      gidx  = (int*)A((size_t)B * 2048 * 32 * sizeof(int));
  float*    dmin  = (float*)A((size_t)B * N0 * sizeof(float));
  float*    attb  = (float*)A((size_t)B * N0 * sizeof(float));
  Ctx ctx{stream, fbuf, fbuf2, stats, wbuf};
  hipStream_t st = stream;

  // original image -> XT f16
  _Float16* imgxt = (_Float16*)A((size_t)B * 53248 * 32 * sizeof(_Float16));
  (void)hipMemsetAsync(imgxt, 0, (size_t)B * 53248 * 32 * sizeof(_Float16), st);
  k_img_to_xt<<<(int)cdivl((long)B * 53248 * 3, 256), 256, 0, st>>>(image, imgxt, 53248, 53248, 3, 32, B);

  // -------- SA stages + image blocks + fusion -------------------------------
  const float* lxp[5]; lxp[0] = pc;
  const float* xcur = xy;
  _Float16* featpre[4]; _Float16* featF[4]; _Float16* img16[4];
  int SpadArr[4];
  for (int i = 0; i < 4; ++i) {
    int Ni = (i == 0) ? N0 : NPTS[i - 1];
    int S = NPTS[i]; int Spad = pad64(S); SpadArr[i] = Spad;
    k_fps<<<B, 256, 0, st>>>(lxp[i], Ni, S, fidx, dmin);
    float* nx = (float*)A((size_t)B * S * 3 * sizeof(float)); lxp[i + 1] = nx;
    k_gather_rows<<<(int)cdivl((long)B * S * 3, 256), 256, 0, st>>>(lxp[i], fidx, nx, 3, S, Ni, B);
    float* xcn = (float*)A((size_t)B * S * 2 * sizeof(float));
    k_gather_rows<<<(int)cdivl((long)B * S * 2, 256), 256, 0, st>>>(xcur, fidx, xcn, 2, S, Ni, B);
    xcur = xcn;
    // image block i
    int HWo = (HS[i] / 2) * (WS_[i] / 2);
    img16[i] = (_Float16*)A((size_t)B * pad64(HWo) * pad32(IMGC[i + 1]) * sizeof(_Float16));
    h_convblock(ctx, ib_w1[i], ib_g1[i], ib_b1[i], ib_w2[i], IMGC[i], IMGC[i + 1],
                HS[i], WS_[i], (i == 0) ? imgxt : img16[i - 1],
                (i == 0) ? 32 : pad32(IMGC[i]), ping, pong, img16[i], B);
    // multi-scale grouping + shared MLPs + maxpool
    int pcc = PCC[i];
    featpre[i] = (_Float16*)A((size_t)B * Spad * pcc * sizeof(_Float16));
    (void)hipMemsetAsync(featpre[i], 0, (size_t)B * Spad * pcc * sizeof(_Float16), st);
    int Cf = (i == 0) ? 0 : PCC[i - 1];
    int chOff = 0;
    for (int j = 0; j < 2; ++j) {
      int K = NS[j];
      k_ballq<<<(int)cdivl((long)B * S, 256), 256, 0, st>>>(lxp[i], nx, gidx, Ni, S, K,
                                                            RAD2[i][j], B);
      int Cg = 3 + Cf, Cgp = pad32(Cg);
      long rowsG = (long)B * S * K;
      (void)hipMemsetAsync(ping, 0, (size_t)rowsG * Cgp * sizeof(_Float16), st);
      k_group<<<(int)cdivl(rowsG, 256), 256, 0, st>>>(
          lxp[i], nx, gidx, (i == 0) ? (const _Float16*)nullptr : featF[i - 1],
          Ni, S, K, Cf, (i == 0) ? 0 : PCC[i - 1], (i == 0) ? 0 : SpadArr[i - 1],
          Cgp, ping, B);
      const _Float16* Xc = ping; int Cp = Cgp; int Cin = Cg;
      _Float16* dsts[3] = {pong, ping, pong};
      for (int l = 0; l < 3; ++l) {
        int O = MLP[i][j][l];
        h_layer(ctx, saW[i][j][l], saG[i][j][l], saB[i][j][l], O, Cin, Xc, Cp,
                dsts[l], S * K, S * K, B);
        Xc = dsts[l]; Cp = pad32(O); Cin = O;
      }
      int Olast = MLP[i][j][2];
      long tm = (long)B * S * Olast;
      k_maxpool<<<(int)cdivl(tm, 256), 256, 0, st>>>(Xc, S, K, Olast, Cp,
                                                     featpre[i], Spad, pcc, chOff, B);
      chOff += Olast;
    }
    // attention fusion with bilinearly sampled image features
    int Cimg = IMGC[i + 1], CimgP = pad32(Cimg);
    int Hh = HS[i] / 2, Ww = WS_[i] / 2;
    _Float16* imgF = (_Float16*)A((size_t)B * Spad * CimgP * sizeof(_Float16));
    (void)hipMemsetAsync(imgF, 0, (size_t)B * Spad * CimgP * sizeof(_Float16), st);
    k_bilinear<<<(int)cdivl((long)B * S, 256), 256, 0, st>>>(
        img16[i], Hh, Ww, pad64(Hh * Ww), xcur, S, Spad, Cimg, CimgP, imgF, B);
    _Float16* imgf_t = (_Float16*)A((size_t)B * Spad * pcc * sizeof(_Float16));
    _Float16* cat_t  = (_Float16*)A((size_t)B * Spad * 2 * pcc * sizeof(_Float16));
    featF[i] = (_Float16*)A((size_t)B * Spad * pcc * sizeof(_Float16));
    h_attention(ctx, fus[i], pcc, Cimg, Spad, S, B, featpre[i], imgF, imgf_t, cat_t,
                attb, featF[i]);
  }

  // -------- FP (feature propagation) stages ---------------------------------
  const _Float16* lvlF[5] = {nullptr, featF[0], featF[1], featF[2], featF[3]};
  int lvlC[5] = {0, 96, 256, 512, 1024};
  int lvlRows[5] = {0, SpadArr[0], SpadArr[1], SpadArr[2], SpadArr[3]};
  int lvlN[5] = {N0, 2048, 512, 128, 32};
  for (int kk = 3; kk >= 0; --kk) {
    int Nu = lvlN[kk];
    int Sk = lvlN[kk + 1];
    int C1 = lvlC[kk];
    int C2 = lvlC[kk + 1];
    int Cp = C1 + C2;
    long rows = (long)B * Nu;
    if (C1 > 0)
      k_copycols16<<<(int)cdivl(rows * C1, 256), 256, 0, st>>>(lvlF[kk], C1, ping, Cp, 0, C1, rows);
    k_interp3<<<(int)cdivl(rows, 256), 256, 0, st>>>(lxp[kk], lxp[kk + 1], Nu, Sk,
                                                     lvlF[kk + 1], lvlRows[kk + 1],
                                                     C2, C2, ping, Cp, C1, B);
    _Float16* nf = (_Float16*)A((size_t)B * Nu * FPM[kk][1] * sizeof(_Float16));
    h_layer(ctx, fpW[kk][0], fpG[kk][0], fpB[kk][0], FPM[kk][0], Cp, ping, Cp, pong, Nu, Nu, B);
    h_layer(ctx, fpW[kk][1], fpG[kk][1], fpB[kk][1], FPM[kk][1], FPM[kk][0], pong,
            pad32(FPM[kk][0]), nf, Nu, Nu, B);
    lvlF[kk] = nf; lvlC[kk] = FPM[kk][1]; lvlRows[kk] = Nu;
  }

  // -------- deconv concat + image fusion conv + final fusion ----------------
  _Float16* dec16 = (_Float16*)A((size_t)B * 53248 * 64 * sizeof(_Float16));
  for (int j = 0; j < 4; ++j) {
    int K = DK[j]; int Hj = HS[j] / 2, Wj = WS_[j] / 2;
    long tot = (long)B * HH * WW * 16;
    k_deconv<<<(int)cdivl(tot, 256), 256, 0, st>>>(img16[j], Hj, Wj, pad64(Hj * Wj),
                                                   IMGC[j + 1], pad32(IMGC[j + 1]),
                                                   dec_w[j], dec_b[j], K, dec16, 53248,
                                                   j * 16, B);
  }
  int Op = h_gemm(ctx, ifc_w, 32, 64, dec16, 64, B * 53248, fbuf);
  _Float16* imf16 = (_Float16*)A((size_t)B * 53248 * 32 * sizeof(_Float16));
  h_bnrelu(ctx, fbuf, Op, 32, ifc_g, ifc_beta, imf16, 53248, 53248, B);
  _Float16* imgFf = (_Float16*)A((size_t)B * N0 * 32 * sizeof(_Float16));
  k_bilinear<<<(int)cdivl((long)B * N0, 256), 256, 0, st>>>(imf16, HH, WW, 53248, xy,
                                                            N0, N0, 32, 32, imgFf, B);
  _Float16* imgf2 = (_Float16*)A((size_t)B * N0 * 128 * sizeof(_Float16));
  _Float16* cat2  = (_Float16*)A((size_t)B * N0 * 256 * sizeof(_Float16));
  _Float16* out16 = (_Float16*)A((size_t)B * N0 * 128 * sizeof(_Float16));
  h_attention(ctx, ffp, 128, 32, N0, N0, B, lvlF[0], imgFf, imgf2, cat2, attb, out16);

  // -------- outputs: (xyz, out[B,128,N]) concatenated flat ------------------
  float* dout = (float*)d_out;
  k_copyf<<<(int)cdivl((long)B * N0 * 3, 256), 256, 0, st>>>(pc, dout, (long)B * N0 * 3);
  k_out_bcn<<<(int)cdivl((long)B * 128 * N0, 256), 256, 0, st>>>(out16, dout + (size_t)B * N0 * 3,
                                                                 N0, 128, 128, B);
}
